// MolecularGraphTransformer_32186484916937
// MI455X (gfx1250) — matrix operand order repaired and checked by reading, not compiled, not run on hardware
//
#include <hip/hip_runtime.h>

#define NNODES 32768
#define NEDGES 131072
#define NHEADS 6
#define CDIM   128
#define HC     768      // NHEADS * CDIM
#define NLAYER 6
#define GSUPER 2048

typedef __attribute__((ext_vector_type(16))) __bf16 bf16x16;
typedef __attribute__((ext_vector_type(8)))  __bf16 bf16x8;
typedef __attribute__((ext_vector_type(8)))  float  floatx8;

// ---------------------------------------------------------------------------
// helpers
// ---------------------------------------------------------------------------
__device__ __forceinline__ float atomic_max_f32(float* addr, float value) {
    int* ai = (int*)addr;
    int old = *ai;
    while (true) {
        float f = __int_as_float(old);
        if (f >= value) break;
        int assumed = old;
        old = atomicCAS(ai, assumed, __float_as_int(value));
        if (old == assumed) break;
    }
    return __int_as_float(old);
}

// ---------------------------------------------------------------------------
// x = embd[x_atoms]  (f32 + bf16 mirror)
// ---------------------------------------------------------------------------
__global__ void embed_kernel(const int* __restrict__ atoms,
                             const float* __restrict__ embd,
                             float* __restrict__ xf,
                             __bf16* __restrict__ xb) {
    int i = blockIdx.x * blockDim.x + threadIdx.x;
    if (i >= NNODES * CDIM) return;
    int n = i >> 7;        // / CDIM
    int c = i & (CDIM - 1);
    float v = embd[atoms[n] * CDIM + c];
    xf[i] = v;
    xb[i] = (__bf16)v;
}

__global__ void f2bf_kernel(const float* __restrict__ in,
                            __bf16* __restrict__ out, int n) {
    int i = blockIdx.x * blockDim.x + threadIdx.x;
    if (i < n) out[i] = (__bf16)in[i];
}

// ---------------------------------------------------------------------------
// Block-tiled WMMA GEMM:  out[m, j] = sum_c X[m,c] * W[j,c] + bias[j]
//   X bf16 [M,128] row-major, W bf16 [J,128] row-major, K = 128.
// One 256-thread block (8 wave32) computes a 128x128 output tile.
// A panel (128x128) and B panel (128x128) are staged once in LDS (32KB+32KB);
// both are contiguous 32KB regions of X / W since K == row length == 128.
// Each wave owns a 16-row M strip: its 4 A fragments (one per K-step) are
// loaded from LDS into registers once and reused across all 8 J-tiles
// -> 32 v_wmma_f32_16x16x32_bf16 per wave, B read from LDS not L2.
// Output mode / relu are template params so the epilogue is branch-free.
// Requires M % 128 == 0 and J % 128 == 0 (true for all call sites).
// ---------------------------------------------------------------------------
template <bool WRITE_F32, bool WRITE_BF16, bool RELU>
__global__ void wmma_gemm(const __bf16* __restrict__ X,
                          const __bf16* __restrict__ W,
                          const float* __restrict__ bias,
                          float* __restrict__ outF,
                          __bf16* __restrict__ outB,
                          int J, int jBlocks) {
    __shared__ __bf16 As[128 * 128];   // 32 KB
    __shared__ __bf16 Bs[128 * 128];   // 32 KB

    int bm = blockIdx.x / jBlocks;
    int bj = blockIdx.x - bm * jBlocks;
    int t  = threadIdx.x;              // 0..255

    // ---- stage panels: each thread copies 8x16B for A and for B ----
    const __bf16* Ablk = X + (size_t)bm * 128 * CDIM;   // contiguous 32KB
    const __bf16* Bblk = W + (size_t)bj * 128 * CDIM;   // contiguous 32KB
#pragma unroll
    for (int i = 0; i < 8; ++i) {
        int off = (t + i * 256) * 8;   // element offset, 8 bf16 per chunk
        *(bf16x8*)(As + off) = *(const bf16x8*)(Ablk + off);
        *(bf16x8*)(Bs + off) = *(const bf16x8*)(Bblk + off);
    }
    __syncthreads();

    int w    = t >> 5;                 // wave id 0..7 -> M strip
    int lane = t & 31;
    int half = lane >> 4;              // K half selector
    int l15  = lane & 15;

    // ---- A fragments for this wave's 16-row strip, all 4 K-steps ----
    const __bf16* Arow = As + (w * 16 + l15) * CDIM;
    bf16x16 afr[4];
#pragma unroll
    for (int ki = 0; ki < 4; ++ki) {
        int k0 = ki * 32;
        bf16x8 alo = *(const bf16x8*)(Arow + k0 + half * 16);
        bf16x8 ahi = *(const bf16x8*)(Arow + k0 + half * 16 + 8);
        afr[ki] = __builtin_shufflevector(alo, ahi,
            0, 1, 2, 3, 4, 5, 6, 7, 8, 9, 10, 11, 12, 13, 14, 15);
    }

    // ---- 8 J-tiles, reusing A fragments ----
#pragma unroll
    for (int tj = 0; tj < 8; ++tj) {
        const __bf16* Brow = Bs + (tj * 16 + l15) * CDIM;
        floatx8 acc = {};
#pragma unroll
        for (int ki = 0; ki < 4; ++ki) {
            bf16x16 b = *(const bf16x16*)(Brow + ki * 32 + half * 16);
            acc = __builtin_amdgcn_wmma_f32_16x16x32_bf16(
                false, afr[ki], false, b, (short)0, acc, false, false);
        }
        int j = bj * 128 + tj * 16 + l15;
        float bjv = bias[j];
#pragma unroll
        for (int r = 0; r < 8; ++r) {
            float v = acc[r] + bjv;
            if (RELU) v = fmaxf(v, 0.f);
            size_t row = (size_t)(bm * 128 + w * 16 + r + 8 * half);
            if (WRITE_F32)  outF[row * J + j] = v;
            if (WRITE_BF16) outB[row * J + j] = (__bf16)v;
        }
    }
}

// ---------------------------------------------------------------------------
// attention buffers init
// ---------------------------------------------------------------------------
__global__ void init_attn(float* __restrict__ agg, float* __restrict__ smax,
                          float* __restrict__ den, int nAgg, int nNH) {
    int i = blockIdx.x * blockDim.x + threadIdx.x;
    if (i < nAgg) agg[i] = 0.f;
    if (i < nNH) { smax[i] = -3.0e38f; den[i] = 0.f; }
}

// ---------------------------------------------------------------------------
// score[e,h] = dot(q[dst,h,:], k[src,h,:]) / sqrt(C);  atomic max into smax[dst,h]
// one wave per edge; 4 channels per lane
// ---------------------------------------------------------------------------
__global__ void edge_score(const int* __restrict__ ei,
                           const __bf16* __restrict__ q,
                           const __bf16* __restrict__ k,
                           float* __restrict__ score,
                           float* __restrict__ smax) {
    int e = (int)((blockIdx.x * (unsigned)blockDim.x + threadIdx.x) >> 5);
    if (e >= NEDGES) return;
    int lane = threadIdx.x & 31;
    int src = ei[e];
    int dst = ei[NEDGES + e];
    const __bf16* qd = q + (size_t)dst * HC;
    const __bf16* ks = k + (size_t)src * HC;
#pragma unroll
    for (int h = 0; h < NHEADS; ++h) {
        int base = h * CDIM + lane * 4;
        float p = (float)qd[base + 0] * (float)ks[base + 0]
                + (float)qd[base + 1] * (float)ks[base + 1]
                + (float)qd[base + 2] * (float)ks[base + 2]
                + (float)qd[base + 3] * (float)ks[base + 3];
#pragma unroll
        for (int off = 16; off >= 1; off >>= 1) p += __shfl_xor(p, off, 32);
        if (lane == 0) {
            float s = p * 0.08838834764831845f;   // 1/sqrt(128)
            score[(size_t)e * NHEADS + h] = s;
            atomic_max_f32(&smax[(size_t)dst * NHEADS + h], s);
        }
    }
}

// ex[e,h] = exp(score - smax[dst,h]);  atomic add into den[dst,h]
__global__ void edge_exp(const int* __restrict__ ei,
                         const float* __restrict__ score,
                         const float* __restrict__ smax,
                         float* __restrict__ exb,
                         float* __restrict__ den) {
    int i = blockIdx.x * blockDim.x + threadIdx.x;
    if (i >= NEDGES * NHEADS) return;
    int e = i / NHEADS;
    int h = i - e * NHEADS;
    int dst = ei[NEDGES + e];
    float ex = __expf(score[i] - smax[(size_t)dst * NHEADS + h]);
    exb[i] = ex;
    atomicAdd(&den[(size_t)dst * NHEADS + h], ex);
}

// agg[dst,c] += (1/H) * sum_h (ex/den) * v[src, h*C + c];  one wave per edge
__global__ void edge_agg(const int* __restrict__ ei,
                         const float* __restrict__ exb,
                         const float* __restrict__ den,
                         const __bf16* __restrict__ v,
                         float* __restrict__ agg) {
    int e = (int)((blockIdx.x * (unsigned)blockDim.x + threadIdx.x) >> 5);
    if (e >= NEDGES) return;
    int lane = threadIdx.x & 31;
    int src = ei[e];
    int dst = ei[NEDGES + e];
    int c4 = lane * 4;
    float a0 = 0.f, a1 = 0.f, a2 = 0.f, a3 = 0.f;
#pragma unroll
    for (int h = 0; h < NHEADS; ++h) {
        float al = exb[(size_t)e * NHEADS + h] / den[(size_t)dst * NHEADS + h];
        const __bf16* vp = v + (size_t)src * HC + h * CDIM + c4;
        a0 += al * (float)vp[0];
        a1 += al * (float)vp[1];
        a2 += al * (float)vp[2];
        a3 += al * (float)vp[3];
    }
    const float invH = 1.0f / NHEADS;
    float* ap = agg + (size_t)dst * CDIM + c4;
    atomicAdd(ap + 0, a0 * invH);
    atomicAdd(ap + 1, a1 * invH);
    atomicAdd(ap + 2, a2 * invH);
    atomicAdd(ap + 3, a3 * invH);
}

// ---------------------------------------------------------------------------
// x_out = relu(LN(agg + skip + x_in));  one wave per node
// ---------------------------------------------------------------------------
__global__ void ln_relu(const float* __restrict__ agg,
                        const float* __restrict__ skip,
                        const float* __restrict__ xin,
                        const float* __restrict__ g,
                        const float* __restrict__ b,
                        float* __restrict__ xoutF,
                        __bf16* __restrict__ xoutB) {
    int n = (int)((blockIdx.x * (unsigned)blockDim.x + threadIdx.x) >> 5);
    if (n >= NNODES) return;
    int lane = threadIdx.x & 31;
    int c4 = lane * 4;
    size_t base = (size_t)n * CDIM + c4;
    float hv[4];
    float s = 0.f, s2 = 0.f;
#pragma unroll
    for (int i = 0; i < 4; ++i) {
        float t = agg[base + i] + skip[base + i] + xin[base + i];
        hv[i] = t; s += t; s2 += t * t;
    }
#pragma unroll
    for (int off = 16; off >= 1; off >>= 1) {
        s  += __shfl_xor(s, off, 32);
        s2 += __shfl_xor(s2, off, 32);
    }
    float mu  = s  * (1.0f / CDIM);
    float var = s2 * (1.0f / CDIM) - mu * mu;
    float rs  = rsqrtf(var + 1e-5f);
#pragma unroll
    for (int i = 0; i < 4; ++i) {
        float y = (hv[i] - mu) * rs * g[c4 + i] + b[c4 + i];
        y = fmaxf(y, 0.f);
        xoutF[base + i] = y;
        xoutB[base + i] = (__bf16)y;
    }
}

// out[g] = dot(m[g,:], Wprop) + bprop;  one wave per supernode
__global__ void prop_out(const __bf16* __restrict__ m,
                         const float* __restrict__ Wp,
                         const float* __restrict__ bp,
                         float* __restrict__ out) {
    int gI = (int)((blockIdx.x * (unsigned)blockDim.x + threadIdx.x) >> 5);
    if (gI >= GSUPER) return;
    int lane = threadIdx.x & 31;
    int c4 = lane * 4;
    const __bf16* mp = m + (size_t)gI * CDIM + c4;
    float p = (float)mp[0] * Wp[c4 + 0] + (float)mp[1] * Wp[c4 + 1]
            + (float)mp[2] * Wp[c4 + 2] + (float)mp[3] * Wp[c4 + 3];
#pragma unroll
    for (int off = 16; off >= 1; off >>= 1) p += __shfl_xor(p, off, 32);
    if (lane == 0) out[gI] = p + bp[0];
}

// ---------------------------------------------------------------------------
// launch
// ---------------------------------------------------------------------------
extern "C" void kernel_launch(void* const* d_in, const int* in_sizes, int n_in,
                              void* d_out, int out_size, void* d_ws, size_t ws_size,
                              hipStream_t stream) {
    (void)in_sizes; (void)n_in; (void)out_size; (void)ws_size;
    const int*   x_atoms = (const int*)d_in[0];
    const int*   eindex  = (const int*)d_in[1];
    const float* embd_w  = (const float*)d_in[2];
    const float* Wq = (const float*)d_in[3];
    const float* bq = (const float*)d_in[4];
    const float* Wk = (const float*)d_in[5];
    const float* bk = (const float*)d_in[6];
    const float* Wv = (const float*)d_in[7];
    const float* bv = (const float*)d_in[8];
    const float* Wsk = (const float*)d_in[9];
    const float* bsk = (const float*)d_in[10];
    const float* lng = (const float*)d_in[11];
    const float* lnb = (const float*)d_in[12];
    const float* Wlin = (const float*)d_in[13];
    const float* blin = (const float*)d_in[14];
    const float* Wprop = (const float*)d_in[15];
    const float* bprop = (const float*)d_in[16];
    float* out = (float*)d_out;

    char* p = (char*)d_ws;
    auto carve = [&](size_t bytes) -> char* {
        char* r = p;
        p += (bytes + 255) & ~(size_t)255;
        return r;
    };
    float*  xA    = (float*) carve((size_t)NNODES * CDIM * 4);
    float*  xB    = (float*) carve((size_t)NNODES * CDIM * 4);
    __bf16* xbf   = (__bf16*)carve((size_t)NNODES * CDIM * 2);
    __bf16* qbf   = (__bf16*)carve((size_t)NNODES * HC * 2);
    __bf16* kbf   = (__bf16*)carve((size_t)NNODES * HC * 2);
    __bf16* vbf   = (__bf16*)carve((size_t)NNODES * HC * 2);
    float*  skip  = (float*) carve((size_t)NNODES * CDIM * 4);
    float*  agg   = (float*) carve((size_t)NNODES * CDIM * 4);
    float*  score = (float*) carve((size_t)NEDGES * NHEADS * 4);
    float*  exb   = (float*) carve((size_t)NEDGES * NHEADS * 4);
    float*  smax  = (float*) carve((size_t)NNODES * NHEADS * 4);
    float*  den   = (float*) carve((size_t)NNODES * NHEADS * 4);
    __bf16* wq_bf = (__bf16*)carve((size_t)HC * CDIM * 2);
    __bf16* wk_bf = (__bf16*)carve((size_t)HC * CDIM * 2);
    __bf16* wv_bf = (__bf16*)carve((size_t)HC * CDIM * 2);
    __bf16* wskbf = (__bf16*)carve((size_t)CDIM * CDIM * 2);
    __bf16* wl_bf = (__bf16*)carve((size_t)CDIM * CDIM * 2);
    __bf16* m1    = (__bf16*)carve((size_t)GSUPER * CDIM * 2);
    __bf16* m2    = (__bf16*)carve((size_t)GSUPER * CDIM * 2);

    const int TB = 256;   // 8 wave32 per block

    embed_kernel<<<(NNODES * CDIM + TB - 1) / TB, TB, 0, stream>>>(x_atoms, embd_w, xA, xbf);

    float* xin  = xA;
    float* xout = xB;

    for (int l = 0; l < NLAYER; ++l) {
        const int nw = HC * CDIM;      // 98304
        const int ns = CDIM * CDIM;    // 16384
        f2bf_kernel<<<(nw + TB - 1) / TB, TB, 0, stream>>>(Wq + (size_t)l * nw, wq_bf, nw);
        f2bf_kernel<<<(nw + TB - 1) / TB, TB, 0, stream>>>(Wk + (size_t)l * nw, wk_bf, nw);
        f2bf_kernel<<<(nw + TB - 1) / TB, TB, 0, stream>>>(Wv + (size_t)l * nw, wv_bf, nw);
        f2bf_kernel<<<(ns + TB - 1) / TB, TB, 0, stream>>>(Wsk + (size_t)l * ns, wskbf, ns);

        {   // Q/K/V projections: M=32768, J=768 -> (256 x 6) 128x128 blocks
            int jB = HC / 128;
            int blocks = (NNODES / 128) * jB;
            wmma_gemm<false, true, false><<<blocks, TB, 0, stream>>>(
                xbf, wq_bf, bq + (size_t)l * HC, nullptr, qbf, HC, jB);
            wmma_gemm<false, true, false><<<blocks, TB, 0, stream>>>(
                xbf, wk_bf, bk + (size_t)l * HC, nullptr, kbf, HC, jB);
            wmma_gemm<false, true, false><<<blocks, TB, 0, stream>>>(
                xbf, wv_bf, bv + (size_t)l * HC, nullptr, vbf, HC, jB);
        }
        {   // skip projection: M=32768, J=128 -> 256 blocks
            int jB = CDIM / 128;
            int blocks = (NNODES / 128) * jB;
            wmma_gemm<true, false, false><<<blocks, TB, 0, stream>>>(
                xbf, wskbf, bsk + (size_t)l * CDIM, skip, nullptr, CDIM, jB);
        }
        {
            int nAgg = NNODES * CDIM, nNH = NNODES * NHEADS;
            init_attn<<<(nAgg + TB - 1) / TB, TB, 0, stream>>>(agg, smax, den, nAgg, nNH);
        }
        edge_score<<<(NEDGES * 32 + TB - 1) / TB, TB, 0, stream>>>(eindex, qbf, kbf, score, smax);
        edge_exp<<<(NEDGES * NHEADS + TB - 1) / TB, TB, 0, stream>>>(eindex, score, smax, exb, den);
        edge_agg<<<(NEDGES * 32 + TB - 1) / TB, TB, 0, stream>>>(eindex, exb, den, vbf, agg);
        ln_relu<<<(NNODES * 32 + TB - 1) / TB, TB, 0, stream>>>(
            agg, skip, xin, lng + (size_t)l * CDIM, lnb + (size_t)l * CDIM, xout, xbf);
        float* t = xin; xin = xout; xout = t;
    }

    {   // output head on supernodes (rows 0..G-1 of x): M=2048, J=128
        const int ns = CDIM * CDIM;
        int jB = CDIM / 128;
        int blocks = (GSUPER / 128) * jB;
        f2bf_kernel<<<(ns + TB - 1) / TB, TB, 0, stream>>>(Wlin, wl_bf, ns);
        wmma_gemm<false, true, true><<<blocks, TB, 0, stream>>>(
            xbf, wl_bf, blin, nullptr, m1, CDIM, jB);
        f2bf_kernel<<<(ns + TB - 1) / TB, TB, 0, stream>>>(Wlin + ns, wl_bf, ns);
        wmma_gemm<false, true, true><<<blocks, TB, 0, stream>>>(
            m1, wl_bf, blin + CDIM, nullptr, m2, CDIM, jB);
    }
    prop_out<<<(GSUPER * 32 + TB - 1) / TB, TB, 0, stream>>>(m2, Wprop, bprop, out);
}
